// CrossAttentionModel_58402965291338
// MI455X (gfx1250) — matrix operand order repaired and checked by reading, not compiled, not run on hardware
//
#include <hip/hip_runtime.h>

typedef __bf16 bf16;
typedef __attribute__((ext_vector_type(16))) __bf16 v16bf;
typedef __attribute__((ext_vector_type(8)))  __bf16 v8bf;
typedef __attribute__((ext_vector_type(8)))  float  v8f;
typedef unsigned int u32x4 __attribute__((ext_vector_type(4)));
typedef int          i32x4 __attribute__((ext_vector_type(4)));
typedef int          i32x8 __attribute__((ext_vector_type(8)));

#define B_   16
#define LQ_  1024
#define LK_  2048
#define E_   512
#define F_   256
#define A_   512

static __device__ __forceinline__ v8f vzero8() {
  v8f z = {0.f,0.f,0.f,0.f,0.f,0.f,0.f,0.f};
  return z;
}

static __device__ __forceinline__ v8f wmma_bf16(v16bf a, v16bf b, v8f c) {
  return __builtin_amdgcn_wmma_f32_16x16x32_bf16(false, a, false, b, (short)0, c,
                                                 false, false);
}

// A-fragment (16x32 bf16): lane m / m+16 hold row m; K chunks [8h..8h+7] and [16+8h..23+8h]
static __device__ __forceinline__ v16bf load_a_bf16(const bf16* base, int ld) {
  int lane = threadIdx.x & 31;
  int row  = lane & 15;
  int half = lane >> 4;
  const bf16* p = base + (size_t)row * ld + 8 * half;
  union { v16bf v; v8bf h[2]; } u;
  u.h[0] = *(const v8bf*)(p);
  u.h[1] = *(const v8bf*)(p + 16);
  return u.v;
}

// B-fragment (32x16 bf16): lane n / n+16 hold column n; K chunks [8h..8h+7] and [16+8h..23+8h]
static __device__ __forceinline__ v16bf load_b_bf16(const bf16* base, int ld) {
  int lane = threadIdx.x & 31;
  int col  = lane & 15;
  int half = lane >> 4;
  const bf16* p = base + (size_t)col * ld + 8 * half;
  union { v16bf v; v8bf h[2]; } u;
  u.h[0] = *(const v8bf*)(p);
  u.h[1] = *(const v8bf*)(p + 16);
  return u.v;
}

// A-fragment from fp32 activations, converted in-register to bf16.
static __device__ __forceinline__ v16bf load_a_f32(const float* base, int ld) {
  int lane = threadIdx.x & 31;
  int row  = lane & 15;
  int half = lane >> 4;
  const float* p = base + (size_t)row * ld + 8 * half;
  const float4* q0 = (const float4*)(p);
  const float4* q1 = (const float4*)(p + 16);
  float4 x0 = q0[0], x1 = q0[1], y0 = q1[0], y1 = q1[1];
  float t[16] = {x0.x,x0.y,x0.z,x0.w, x1.x,x1.y,x1.z,x1.w,
                 y0.x,y0.y,y0.z,y0.w, y1.x,y1.y,y1.z,y1.w};
  v16bf a;
  #pragma unroll
  for (int j = 0; j < 16; ++j) a[j] = (bf16)t[j];
  return a;
}

// ---------------- TDM: stage a contiguous run of 16-bit elements into LDS ----------------
#if defined(__has_builtin)
#if __has_builtin(__builtin_amdgcn_tensor_load_to_lds)
#define HAVE_TDM 1
#endif
#if __has_builtin(__builtin_amdgcn_s_wait_tensorcnt)
#define HAVE_TENSORCNT 1
#endif
#endif

#ifdef HAVE_TDM
static __device__ __forceinline__ void tdm_load_contig_to_lds(unsigned lds_off,
                                                              const void* g,
                                                              int elems_u16) {
  // D# group0: count=1, type=2 ("image"), lds_addr, global_addr
  unsigned long long ga = (unsigned long long)(size_t)g;
  u32x4 g0;
  g0[0] = 1u;                                            // count=1, user mode
  g0[1] = lds_off;                                       // lds_addr (bytes)
  g0[2] = (unsigned)(ga & 0xffffffffu);                  // global_addr[31:0]
  g0[3] = (unsigned)((ga >> 32) & 0x01ffffffu) | 0x80000000u; // addr[56:32] | type=2
  // D# group1: data_size=2B, 1-D tensor: tensor_dim0 = tile_dim0 = elems, tile_dim1=1
  i32x8 g1;
  g1[0] = 0x00010000;                                    // data_size=1 (2 bytes)
  g1[1] = (int)((unsigned)elems_u16 << 16);              // tensor_dim0[15:0] @bits48-63
  g1[2] = (int)(((unsigned)elems_u16 >> 16) | (1u << 16)); // dim0[31:16] | tensor_dim1=1
  g1[3] = (int)((unsigned)elems_u16 << 16);              // tile_dim0 @bits112-127
  g1[4] = 1;                                             // tile_dim1=1, tile_dim2=0
  g1[5] = elems_u16;                                     // tensor_dim0_stride[31:0]
  g1[6] = 0;
  g1[7] = 0;
  i32x4 gz = {0, 0, 0, 0};
#if defined(__clang_major__) && __clang_major__ >= 23
  i32x8 gz8 = {0, 0, 0, 0, 0, 0, 0, 0};
  __builtin_amdgcn_tensor_load_to_lds(g0, g1, gz, gz, gz8, 0);
#else
  __builtin_amdgcn_tensor_load_to_lds(g0, g1, gz, gz, 0);
#endif
}
#endif

// WT[n*K + k] = (bf16)W[k*N + n]
__global__ void transpose_cvt_kernel(const float* __restrict__ W, bf16* __restrict__ WT,
                                     int K, int N) {
  int idx = blockIdx.x * blockDim.x + threadIdx.x;
  if (idx >= K * N) return;
  int n = idx / K;
  int k = idx - n * K;
  WT[idx] = (bf16)W[(size_t)k * N + n];
}

// C[16x64 tile] = X(fp32)[M,K] @ WT(bf16)[N,K]^T + bias -> bf16
// transposed==0 : out[m*N+n] ; transposed==1 : out[(b*N+n)*rowsPB + key], rowsPB = 1<<bshift
__global__ __launch_bounds__(32)
void proj_gemm_kernel(const float* __restrict__ X, const bf16* __restrict__ WT,
                      const float* __restrict__ bias, bf16* __restrict__ out,
                      int K, int N, int bshift, int transposed) {
  int m0 = blockIdx.x * 16;
  int n0 = blockIdx.y * 64;
  int lane = threadIdx.x & 31;
  int half = lane >> 4, lcol = lane & 15;

  v8f c[4];
  #pragma unroll
  for (int t = 0; t < 4; ++t) c[t] = vzero8();

  auto ldb = [&](v16bf* dst, int kk) {
    #pragma unroll
    for (int t = 0; t < 4; ++t)
      dst[t] = load_b_bf16(WT + (size_t)(n0 + t * 16) * K + kk * 32, K);
  };

  int nk = K >> 5;                      // 16 (E) or 8 (F); always even
  v16bf a0, a1, b0[4], b1[4];
  a0 = load_a_f32(X + (size_t)m0 * K, K);
  ldb(b0, 0);
  for (int kk = 0; kk < nk; kk += 2) {
    a1 = load_a_f32(X + (size_t)m0 * K + (kk + 1) * 32, K);
    ldb(b1, kk + 1);
    #pragma unroll
    for (int t = 0; t < 4; ++t) c[t] = wmma_bf16(a0, b0[t], c[t]);
    if (kk + 2 < nk) {
      a0 = load_a_f32(X + (size_t)m0 * K + (kk + 2) * 32, K);
      ldb(b0, kk + 2);
    }
    #pragma unroll
    for (int t = 0; t < 4; ++t) c[t] = wmma_bf16(a1, b1[t], c[t]);
  }

  int rowsPB = 1 << bshift;
  int bidx = m0 >> bshift;
  int key0 = m0 & (rowsPB - 1);
  #pragma unroll
  for (int t = 0; t < 4; ++t) {
    int n = n0 + t * 16 + lcol;
    float bv = bias[n];
    #pragma unroll
    for (int r = 0; r < 8; ++r) {
      int mr = r + 8 * half;
      float v = c[t][r] + bv;
      if (!transposed)
        out[(size_t)(m0 + mr) * N + n] = (bf16)v;
      else
        out[((size_t)bidx * N + n) * rowsPB + (key0 + mr)] = (bf16)v;
    }
  }
}

// Fused attention + output projection. 1 block = 16 queries of one batch; 8 waves.
__global__ __launch_bounds__(256)
void attn_kernel(const bf16* __restrict__ Qb, const bf16* __restrict__ Kb,
                 const bf16* __restrict__ Vt, const bf16* __restrict__ WoT,
                 const float* __restrict__ bo, float* __restrict__ out) {
  extern __shared__ char smem[];
  bf16* Pbuf   = (bf16*)smem;          // 16 x LK_ bf16 = 64 KB
  bf16* Qstage = (bf16*)smem;          // 16 x A_ bf16 (consumed before Pbuf written)
  __shared__ float red[128];           // 8 waves x 16 rows

  int b    = blockIdx.x >> 6;          // LQ_/16 == 64
  int qt   = blockIdx.x & 63;
  int wave = threadIdx.x >> 5;
  int lane = threadIdx.x & 31;
  int half = lane >> 4, lcol = lane & 15;

  // ---- stage Q tile (16 x A_ bf16, contiguous 16 KB) into LDS via TDM ----
  const bf16* Qt = Qb + ((size_t)b * LQ_ + qt * 16) * A_;
#ifdef HAVE_TDM
  if (wave == 0) {
    tdm_load_contig_to_lds((unsigned)(size_t)(void*)Qstage, (const void*)Qt, 16 * A_);
#ifdef HAVE_TENSORCNT
    __builtin_amdgcn_s_wait_tensorcnt(0);
#endif
  }
#else
  for (int i = threadIdx.x; i < (16 * A_) / 8; i += 256)
    ((uint4*)Qstage)[i] = ((const uint4*)Qt)[i];
#endif
  __syncthreads();

  // warm L2 with the V / Wo rows this wave consumes in later phases
  const bf16* Vw = Vt + (size_t)b * A_ * LK_ + (size_t)(wave * 64) * LK_;
  const bf16* Ww = WoT + (size_t)(wave * 64) * A_;
  #pragma unroll
  for (int rr = 0; rr < 2; ++rr) {
    __builtin_prefetch(Vw + (size_t)(rr * 32 + lane) * LK_, 0, 1);
    __builtin_prefetch(Ww + (size_t)(rr * 32 + lane) * A_, 0, 1);
  }

  // Q A-fragments from LDS into registers (reused 16x each)
  v16bf qa[16];
  #pragma unroll
  for (int kk = 0; kk < 16; ++kk) qa[kk] = load_a_bf16(Qstage + kk * 32, A_);

  // ---- scores: wave covers 256 keys (16 N-tiles); double-buffered 4-frag chunks ----
  const bf16* Kw = Kb + (size_t)b * LK_ * A_ + (size_t)(wave * 256) * A_;
  auto ldk = [&](v16bf* dst, int cc) {
    int f_ = cc >> 2, kb_ = (cc & 3) * 4;
    #pragma unroll
    for (int j = 0; j < 4; ++j)
      dst[j] = load_b_bf16(Kw + (size_t)(f_ * 16) * A_ + (kb_ + j) * 32, A_);
  };

  v8f sf[16];
  {
    v16bf buf0[4], buf1[4];
    ldk(buf0, 0);
    v8f c = vzero8();
    #pragma unroll
    for (int cc = 0; cc < 64; ++cc) {
      v16bf* cur = (cc & 1) ? buf1 : buf0;
      v16bf* nxt = (cc & 1) ? buf0 : buf1;
      if (cc < 63) ldk(nxt, cc + 1);
      int kb = (cc & 3) * 4;
      #pragma unroll
      for (int j = 0; j < 4; ++j) c = wmma_bf16(qa[kb + j], cur[j], c);
      if ((cc & 3) == 3) { sf[cc >> 2] = c; c = vzero8(); }
    }
  }

  // ---- softmax: row max (shfl_xor within wave, LDS across waves) ----
  float mrow[8];
  #pragma unroll
  for (int r = 0; r < 8; ++r) {
    float m = sf[0][r];
    #pragma unroll
    for (int f = 1; f < 16; ++f) m = fmaxf(m, sf[f][r]);
    #pragma unroll
    for (int s = 1; s < 16; s <<= 1) m = fmaxf(m, __shfl_xor(m, s, 32));
    mrow[r] = m;
  }
  if (lcol == 0) {
    #pragma unroll
    for (int r = 0; r < 8; ++r) red[wave * 16 + r + 8 * half] = mrow[r];
  }
  __syncthreads();
  #pragma unroll
  for (int r = 0; r < 8; ++r) {
    float m = -3.402823466e38f;
    #pragma unroll
    for (int w = 0; w < 8; ++w) m = fmaxf(m, red[w * 16 + r + 8 * half]);
    mrow[r] = m;
  }
  __syncthreads();

  float srow[8];
  #pragma unroll
  for (int r = 0; r < 8; ++r) {
    float s = 0.f;
    #pragma unroll
    for (int f = 0; f < 16; ++f) {
      float e = __expf(sf[f][r] - mrow[r]);
      sf[f][r] = e;
      s += e;
    }
    #pragma unroll
    for (int t = 1; t < 16; t <<= 1) s += __shfl_xor(s, t, 32);
    srow[r] = s;
  }
  if (lcol == 0) {
    #pragma unroll
    for (int r = 0; r < 8; ++r) red[wave * 16 + r + 8 * half] = srow[r];
  }
  __syncthreads();
  #pragma unroll
  for (int r = 0; r < 8; ++r) {
    float s = 0.f;
    #pragma unroll
    for (int w = 0; w < 8; ++w) s += red[w * 16 + r + 8 * half];
    srow[r] = 1.f / s;                 // fold normalization into P
  }

  // normalized P -> LDS (bf16, row-major 16 x LK_)
  #pragma unroll
  for (int f = 0; f < 16; ++f) {
    int n = wave * 256 + f * 16 + lcol;
    #pragma unroll
    for (int r = 0; r < 8; ++r)
      Pbuf[(size_t)(r + 8 * half) * LK_ + n] = (bf16)(sf[f][r] * srow[r]);
  }
  __syncthreads();

  // ---- context: O = P @ V ; wave owns 64 A-columns, all 2048 keys; double-buffered ----
  auto ldv = [&](v16bf* dst, int kk) {
    #pragma unroll
    for (int t = 0; t < 4; ++t)
      dst[t] = load_b_bf16(Vw + (size_t)(t * 16) * LK_ + kk * 32, LK_);
  };
  v8f oc[4];
  #pragma unroll
  for (int t = 0; t < 4; ++t) oc[t] = vzero8();
  {
    v16bf vb0[4], vb1[4], pa0, pa1;
    ldv(vb0, 0);
    pa0 = load_a_bf16(Pbuf, LK_);
    #pragma unroll
    for (int kk = 0; kk < 64; ++kk) {
      v16bf* cur = (kk & 1) ? vb1 : vb0;
      v16bf* nxt = (kk & 1) ? vb0 : vb1;
      v16bf  pac = (kk & 1) ? pa1 : pa0;
      if (kk < 63) {
        ldv(nxt, kk + 1);
        if (kk & 1) pa0 = load_a_bf16(Pbuf + (kk + 1) * 32, LK_);
        else        pa1 = load_a_bf16(Pbuf + (kk + 1) * 32, LK_);
      }
      #pragma unroll
      for (int t = 0; t < 4; ++t) oc[t] = wmma_bf16(pac, cur[t], oc[t]);
    }
  }
  __syncthreads();                     // all P reads complete before aliasing Pbuf

  // stage O (bf16) in LDS for the final projection's A-fragments
  bf16* Obuf = (bf16*)smem;            // 16 x A_
  #pragma unroll
  for (int t = 0; t < 4; ++t) {
    int n = wave * 64 + t * 16 + lcol;
    #pragma unroll
    for (int r = 0; r < 8; ++r)
      Obuf[(size_t)(r + 8 * half) * A_ + n] = (bf16)oc[t][r];
  }
  __syncthreads();

  // ---- out = O @ Wo + bo ; wave owns 64 E-columns; double-buffered ----
  auto ldw = [&](v16bf* dst, int kk) {
    #pragma unroll
    for (int t = 0; t < 4; ++t)
      dst[t] = load_b_bf16(Ww + (size_t)(t * 16) * A_ + kk * 32, A_);
  };
  v8f fc[4];
  #pragma unroll
  for (int t = 0; t < 4; ++t) fc[t] = vzero8();
  {
    v16bf wb0[4], wb1[4];
    ldw(wb0, 0);
    #pragma unroll
    for (int kk = 0; kk < 16; ++kk) {
      v16bf* cur = (kk & 1) ? wb1 : wb0;
      v16bf* nxt = (kk & 1) ? wb0 : wb1;
      if (kk < 15) ldw(nxt, kk + 1);
      v16bf oa = load_a_bf16(Obuf + kk * 32, A_);
      #pragma unroll
      for (int t = 0; t < 4; ++t) fc[t] = wmma_bf16(oa, cur[t], fc[t]);
    }
  }
  float* Orow = out + ((size_t)b * LQ_ + qt * 16) * E_;
  #pragma unroll
  for (int t = 0; t < 4; ++t) {
    int n = wave * 64 + t * 16 + lcol;
    float bb = bo[n];
    #pragma unroll
    for (int r = 0; r < 8; ++r)
      Orow[(size_t)(r + 8 * half) * E_ + n] = fc[t][r] + bb;
  }
}

extern "C" void kernel_launch(void* const* d_in, const int* in_sizes, int n_in,
                              void* d_out, int out_size, void* d_ws, size_t ws_size,
                              hipStream_t stream) {
  const float* X  = (const float*)d_in[0];   // [B,LQ,E]
  const float* Yf = (const float*)d_in[1];   // [B,LK,F]
  const float* Wq = (const float*)d_in[2];
  const float* bq = (const float*)d_in[3];
  const float* Wk = (const float*)d_in[4];
  const float* bk = (const float*)d_in[5];
  const float* Wv = (const float*)d_in[6];
  const float* bv = (const float*)d_in[7];
  const float* Wo = (const float*)d_in[8];
  const float* bo = (const float*)d_in[9];
  float* out = (float*)d_out;

  char* ws = (char*)d_ws;
  size_t off = 0;
  auto take = [&](size_t elems) {
    bf16* p = (bf16*)(ws + off);
    off += ((elems * sizeof(bf16) + 255) & ~(size_t)255);
    return p;
  };
  bf16* WqT = take((size_t)A_ * E_);          // [A][E]
  bf16* WkT = take((size_t)A_ * F_);          // [A][F]
  bf16* WvT = take((size_t)A_ * F_);          // [A][F]
  bf16* WoT = take((size_t)E_ * A_);          // [E][A]
  bf16* Qb  = take((size_t)B_ * LQ_ * A_);    // row-major
  bf16* Kb  = take((size_t)B_ * LK_ * A_);    // row-major
  bf16* Vt  = take((size_t)B_ * A_ * LK_);    // transposed per batch

  transpose_cvt_kernel<<<(E_ * A_ + 255) / 256, 256, 0, stream>>>(Wq, WqT, E_, A_);
  transpose_cvt_kernel<<<(F_ * A_ + 255) / 256, 256, 0, stream>>>(Wk, WkT, F_, A_);
  transpose_cvt_kernel<<<(F_ * A_ + 255) / 256, 256, 0, stream>>>(Wv, WvT, F_, A_);
  transpose_cvt_kernel<<<(A_ * E_ + 255) / 256, 256, 0, stream>>>(Wo, WoT, A_, E_);

  dim3 gq(B_ * LQ_ / 16, A_ / 64);   // 1024 x 8
  proj_gemm_kernel<<<gq, 32, 0, stream>>>(X,  WqT, bq, Qb, E_, A_, 10, 0);
  dim3 gk(B_ * LK_ / 16, A_ / 64);   // 2048 x 8
  proj_gemm_kernel<<<gk, 32, 0, stream>>>(Yf, WkT, bk, Kb, F_, A_, 11, 0);
  proj_gemm_kernel<<<gk, 32, 0, stream>>>(Yf, WvT, bv, Vt, F_, A_, 11, 1);

  attn_kernel<<<B_ * (LQ_ / 16), 256, (size_t)16 * LK_ * sizeof(bf16), stream>>>(
      Qb, Kb, Vt, WoT, bo, out);
}
